// GNNEdgePredictor_85744727097862
// MI455X (gfx1250) — compile-verified, matrix operand order checked by reading, not run on hardware
//
#include <hip/hip_runtime.h>
#include <hip/hip_bf16.h>
#include <math.h>

typedef __attribute__((ext_vector_type(16))) __bf16 v16bf;
typedef __attribute__((ext_vector_type(8)))  __bf16 v8bf;
typedef __attribute__((ext_vector_type(4)))  __bf16 v4bf;
typedef __attribute__((ext_vector_type(8)))  float  v8f;

#define NND 100000
#define NED 1600000
#define DF  128
#define LDP 136   // padded LDS row stride (elements): 272 B = 16B-aligned, bank +4/row

// Load one 16x16x32 bf16 fragment (A-style or transposed-B-style):
// lane's 16 values = two contiguous 8-element (16 B) runs of the row.
__device__ __forceinline__ v16bf ldfrag(const __bf16* rowk, int half) {
  v8bf lo = *(const v8bf*)(rowk + half * 8);
  v8bf hi = *(const v8bf*)(rowk + 16 + half * 8);
  return __builtin_shufflevector(lo, hi, 0, 1, 2, 3, 4, 5, 6, 7,
                                 8, 9, 10, 11, 12, 13, 14, 15);
}

// ---------------- small utility kernels ----------------
__global__ void k_fill1(float* p, int n) {
  int i = blockIdx.x * 256 + threadIdx.x;
  if (i < n) p[i] = 1.0f;
}

__global__ void k_degcount(const int* __restrict__ dst, float* __restrict__ deg) {
  int e = blockIdx.x * 256 + threadIdx.x;
  if (e < NED) atomicAdd(&deg[dst[e]], 1.0f);
}

__global__ void k_rsqrt(float* p, int n) {
  int i = blockIdx.x * 256 + threadIdx.x;
  if (i < n) p[i] = rsqrtf(p[i]);
}

// wm1t[n*256 + k] = (bf16) Wm1[k*128 + n]   (transpose + convert, one-time)
__global__ void k_cvt_wm1t(const float* __restrict__ Wm1, __bf16* __restrict__ wm1t) {
  int o = blockIdx.x * 256 + threadIdx.x;
  if (o < 256 * 128) {
    int n = o >> 8, k = o & 255;
    wm1t[o] = (__bf16)Wm1[k * 128 + n];
  }
}

// B[i] = A[i] * dinv[row]^2   (self-loop contribution initializes the agg buffer)
__global__ void k_agg_init(const float* __restrict__ A, const float* __restrict__ dinv,
                           float* __restrict__ B) {
  int i = blockIdx.x * 256 + threadIdx.x;
  if (i < NND * DF) {
    float di = dinv[i >> 7];
    B[i] = A[i] * di * di;
  }
}

// per-edge scatter: B[dst] += A[src] * dinv[src]*dinv[dst]; 32 lanes/edge, 4 floats/lane
__global__ void k_scatter(const float* __restrict__ A, const float* __restrict__ dinv,
                          const int* __restrict__ src, const int* __restrict__ dst,
                          float* __restrict__ B) {
  int t = blockIdx.x * 256 + threadIdx.x;
  int e = t >> 5;
  int f4 = (t & 31) * 4;
  if (e < NED) {
    int s = src[e], d = dst[e];
    float w = dinv[s] * dinv[d];
    const float4 hv = *(const float4*)&A[(size_t)s * DF + f4];
    float* bp = &B[(size_t)d * DF + f4];
    atomicAdd(bp + 0, hv.x * w);
    atomicAdd(bp + 1, hv.y * w);
    atomicAdd(bp + 2, hv.z * w);
    atomicAdd(bp + 3, hv.w * w);
  }
}

// h = relu(B + bias); optional fp32 and bf16 outputs
__global__ void k_final(const float* __restrict__ B, const float* __restrict__ bias,
                        float* __restrict__ Hout, __bf16* __restrict__ Hbf) {
  int i = blockIdx.x * 256 + threadIdx.x;
  if (i < NND * DF) {
    float v = B[i] + bias[i & 127];
    v = v > 0.0f ? v : 0.0f;
    if (Hout) Hout[i] = v;
    if (Hbf)  Hbf[i]  = (__bf16)v;
  }
}

// ---------------- node GEMM: Out[64 x 128] = X[64 x 128] @ W[128 x 128] ----------------
__global__ __launch_bounds__(128) void k_gemm_node(const float* __restrict__ X,
                                                   const float* __restrict__ W,
                                                   float* __restrict__ Out, int nrows) {
  __shared__ __align__(16) __bf16 Wlt[128 * LDP];  // W transposed: Wlt[n*LDP + k]
  __shared__ __align__(16) __bf16 Al[64 * LDP];    // Al[row*LDP + k]
  const int tid = threadIdx.x;
  const int rowBase = blockIdx.x * 64;

  // stage W transposed: read float4 along n (coalesced), scatter 4 bf16 rows in LDS
  for (int c = tid; c < 128 * 32; c += 128) {       // 4096 chunks of 4 elements
    int k = c >> 5, n4 = (c & 31) * 4;
    float4 wv = *(const float4*)&W[k * 128 + n4];
    Wlt[(n4 + 0) * LDP + k] = (__bf16)wv.x;
    Wlt[(n4 + 1) * LDP + k] = (__bf16)wv.y;
    Wlt[(n4 + 2) * LDP + k] = (__bf16)wv.z;
    Wlt[(n4 + 3) * LDP + k] = (__bf16)wv.w;
  }
  // stage A: float4 read -> 4 bf16 (8 B) store
  for (int c = tid; c < 64 * 32; c += 128) {        // 2048 chunks of 4 elements
    int le = c >> 5, k4 = (c & 31) * 4;
    int r = rowBase + le;
    v4bf bv;
    if (r < nrows) {
      float4 xv = *(const float4*)&X[(size_t)r * DF + k4];
      bv[0] = (__bf16)xv.x; bv[1] = (__bf16)xv.y;
      bv[2] = (__bf16)xv.z; bv[3] = (__bf16)xv.w;
    } else {
      bv[0] = (__bf16)0.0f; bv[1] = (__bf16)0.0f;
      bv[2] = (__bf16)0.0f; bv[3] = (__bf16)0.0f;
    }
    *(v4bf*)&Al[le * LDP + k4] = bv;
  }
  __syncthreads();

  const int wave = tid >> 5, lane = tid & 31;
  const int half = lane >> 4, nl = lane & 15;
  const int m0 = wave * 16;

  v8f acc[8];
#pragma unroll
  for (int nt = 0; nt < 8; ++nt)
#pragma unroll
    for (int j = 0; j < 8; ++j) acc[nt][j] = 0.0f;

#pragma unroll
  for (int kk = 0; kk < 128; kk += 32) {
    v16bf a = ldfrag(&Al[(m0 + nl) * LDP + kk], half);
#pragma unroll
    for (int nt = 0; nt < 8; ++nt) {
      v16bf b = ldfrag(&Wlt[(nt * 16 + nl) * LDP + kk], half);
      acc[nt] = __builtin_amdgcn_wmma_f32_16x16x32_bf16(false, a, false, b,
                                                        (short)0, acc[nt], false, false);
    }
  }

#pragma unroll
  for (int nt = 0; nt < 8; ++nt)
#pragma unroll
    for (int v = 0; v < 8; ++v) {
      int r = rowBase + m0 + v + 8 * half;
      if (r < nrows) Out[(size_t)r * DF + nt * 16 + nl] = acc[nt][v];
    }
}

// ---------------- edge MLP: 64 edges/block, K=256 in two 128-phases (src, dst halves) ---------
__global__ __launch_bounds__(128) void k_edge_mlp(const __bf16* __restrict__ h2bf,
                                                  const __bf16* __restrict__ wm1t,
                                                  const int* __restrict__ src,
                                                  const int* __restrict__ dst,
                                                  const float* __restrict__ bm1,
                                                  const float* __restrict__ wm2,
                                                  const float* __restrict__ bm2,
                                                  float* __restrict__ out) {
  __shared__ __align__(16) __bf16 Wlt[128 * LDP];  // Wm1^T K-half: Wlt[n*LDP + k]
  __shared__ __align__(16) __bf16 Al[64 * LDP];    // gathered features, one K-half
  const int tid = threadIdx.x;
  const int eBase = blockIdx.x * 64;
  const int wave = tid >> 5, lane = tid & 31;
  const int half = lane >> 4, nl = lane & 15;
  const int m0 = wave * 16;

  v8f acc[8];
#pragma unroll
  for (int nt = 0; nt < 8; ++nt)
#pragma unroll
    for (int j = 0; j < 8; ++j) acc[nt][j] = 0.0f;

  for (int p = 0; p < 2; ++p) {
    // stage Wm1^T K-half from pre-transposed global (contiguous 16 B chunks)
    for (int c = tid; c < 128 * 16; c += 128) {
      int n = c >> 4, k8 = (c & 15) * 8;
      *(uint4*)&Wlt[n * LDP + k8] =
          *(const uint4*)&wm1t[(size_t)n * 256 + p * 128 + k8];
    }
    // stage gathered A K-half: src rows for p==0, dst rows for p==1
    const int* __restrict__ idx = p ? dst : src;
    for (int c = tid; c < 64 * 16; c += 128) {
      int le = c >> 4, k8 = (c & 15) * 8;
      int e = eBase + le;
      uint4 val = make_uint4(0u, 0u, 0u, 0u);
      if (e < NED) {
        int r = idx[e];
        val = *(const uint4*)&h2bf[(size_t)r * 128 + k8];
      }
      *(uint4*)&Al[le * LDP + k8] = val;
    }
    __syncthreads();

#pragma unroll
    for (int kk = 0; kk < 128; kk += 32) {
      v16bf a = ldfrag(&Al[(m0 + nl) * LDP + kk], half);
#pragma unroll
      for (int nt = 0; nt < 8; ++nt) {
        v16bf b = ldfrag(&Wlt[(nt * 16 + nl) * LDP + kk], half);
        acc[nt] = __builtin_amdgcn_wmma_f32_16x16x32_bf16(false, a, false, b,
                                                          (short)0, acc[nt], false, false);
      }
    }
    __syncthreads();
  }

  // epilogue: relu(z + bm1) @ Wm2 + bm2, sigmoid.  C/D layout: lane nl = column,
  // VGPR v = row v + 8*half -> reduce over the 16 lanes within each half-wave.
  const float bm2v = bm2[0];
#pragma unroll
  for (int v = 0; v < 8; ++v) {
    float partial = 0.0f;
#pragma unroll
    for (int nt = 0; nt < 8; ++nt) {
      int n = nt * 16 + nl;
      float z = acc[nt][v] + bm1[n];
      z = z > 0.0f ? z : 0.0f;
      partial += z * wm2[n];
    }
    partial += __shfl_xor(partial, 1, 32);
    partial += __shfl_xor(partial, 2, 32);
    partial += __shfl_xor(partial, 4, 32);
    partial += __shfl_xor(partial, 8, 32);
    if (nl == 0) {
      int e = eBase + m0 + v + 8 * half;
      if (e < NED) {
        float logit = partial + bm2v;
        out[e] = 1.0f / (1.0f + expf(-logit));
      }
    }
  }
}

// ---------------- host launcher ----------------
extern "C" void kernel_launch(void* const* d_in, const int* in_sizes, int n_in,
                              void* d_out, int out_size, void* d_ws, size_t ws_size,
                              hipStream_t stream) {
  const float* x   = (const float*)d_in[0];
  const int*   ei  = (const int*)d_in[1];
  const float* W1  = (const float*)d_in[2];
  const float* b1  = (const float*)d_in[3];
  const float* W2  = (const float*)d_in[4];
  const float* b2  = (const float*)d_in[5];
  const float* Wm1 = (const float*)d_in[6];
  const float* bm1 = (const float*)d_in[7];
  const float* Wm2 = (const float*)d_in[8];
  const float* bm2 = (const float*)d_in[9];
  float* out = (float*)d_out;

  const int* src = ei;
  const int* dst = ei + NED;

  // workspace layout (all offsets multiple of 256 bytes)
  char* ws = (char*)d_ws;
  const size_t SZ_NODE = (size_t)NND * DF * sizeof(float);      // 51.2 MB
  const size_t OFF_DINV  = 0;
  const size_t OFF_A     = 524288;
  const size_t OFF_B     = OFF_A + SZ_NODE;
  const size_t OFF_C     = OFF_B + SZ_NODE;
  const size_t OFF_H2BF  = OFF_C + SZ_NODE;
  const size_t OFF_WM1T  = OFF_H2BF + (size_t)NND * DF * sizeof(__bf16);

  float*  dinv  = (float*)(ws + OFF_DINV);
  float*  bufA  = (float*)(ws + OFF_A);
  float*  bufB  = (float*)(ws + OFF_B);
  float*  bufC  = (float*)(ws + OFF_C);
  __bf16* h2bf  = (__bf16*)(ws + OFF_H2BF);
  __bf16* wm1t  = (__bf16*)(ws + OFF_WM1T);

  const int gNode = (NND + 255) / 256;       // 391
  const int gEdge = (NED + 255) / 256;       // 6250
  const int gFeat = (NND * DF) / 256;        // 50000
  const int gScat = (NED * 32) / 256;        // 200000
  const int gGemm = (NND + 63) / 64;         // 1563
  const int gEMLP = (NED + 63) / 64;         // 25000

  // degrees with self loops -> dinv = rsqrt(deg)
  k_fill1   <<<gNode, 256, 0, stream>>>(dinv, NND);
  k_degcount<<<gEdge, 256, 0, stream>>>(dst, dinv);
  k_rsqrt   <<<gNode, 256, 0, stream>>>(dinv, NND);

  // pre-transpose+convert Wm1 -> bf16 [128 n][256 k]
  k_cvt_wm1t<<<(256 * 128 + 255) / 256, 256, 0, stream>>>(Wm1, wm1t);

  // layer 1: h1 = relu(agg(x@W1) + b1)
  k_gemm_node<<<gGemm, 128, 0, stream>>>(x, W1, bufA, NND);
  k_agg_init <<<gFeat, 256, 0, stream>>>(bufA, dinv, bufB);
  k_scatter  <<<gScat, 256, 0, stream>>>(bufA, dinv, src, dst, bufB);
  k_final    <<<gFeat, 256, 0, stream>>>(bufB, b1, bufC, (__bf16*)nullptr);

  // layer 2: h2 = relu(agg(h1@W2) + b2)  (bf16 copy for the edge stage)
  k_gemm_node<<<gGemm, 128, 0, stream>>>(bufC, W2, bufA, NND);
  k_agg_init <<<gFeat, 256, 0, stream>>>(bufA, dinv, bufB);
  k_scatter  <<<gScat, 256, 0, stream>>>(bufA, dinv, src, dst, bufB);
  k_final    <<<gFeat, 256, 0, stream>>>(bufB, b2, (float*)nullptr, h2bf);

  // edge MLP + sigmoid
  k_edge_mlp<<<gEMLP, 128, 0, stream>>>(h2bf, wm1t, src, dst, bm1, Wm2, bm2, out);
}
